// DaRnnModel_5188320493646
// MI455X (gfx1250) — matrix operand order repaired and checked by reading, not compiled, not run on hardware
//
#include <hip/hip_runtime.h>

typedef __bf16 bf16;
typedef __attribute__((ext_vector_type(16))) __bf16 v16bf;
typedef __attribute__((ext_vector_type(8)))  float  v8f;

#define TT 128
#define BB 512
#define NN 256
#define MM 512
#define M2 1024
#define M4 2048

#define WMMA_BF16(acc, a, b) \
  (acc) = __builtin_amdgcn_wmma_f32_16x16x32_bf16(false, (a), false, (b), (short)0, (acc), false, false)

// ---- fast transcendental (gfx1250 has V_TANH_F32) ----
#if __has_builtin(__builtin_amdgcn_tanhf)
#define FAST_TANH(x) __builtin_amdgcn_tanhf((x))
#elif __has_builtin(__builtin_amdgcn_tanh_f32)
#define FAST_TANH(x) __builtin_amdgcn_tanh_f32((x))
#else
#define FAST_TANH(x) tanhf((x))
#endif

// ---- async global->LDS copy (CDNA5 ASYNCcnt path), guarded with fallback ----
typedef int async_v4i __attribute__((vector_size(16)));

#if __has_builtin(__builtin_amdgcn_global_load_async_to_lds_b128)
#define HAVE_ASYNC_LDS 1
#define ASYNC_CP16(dst_lds, src_g)                                        \
  __builtin_amdgcn_global_load_async_to_lds_b128(                         \
      (__attribute__((address_space(1))) async_v4i*)(src_g),              \
      (__attribute__((address_space(3))) async_v4i*)(dst_lds), 0, 0)
#else
#define HAVE_ASYNC_LDS 0
#define ASYNC_CP16(dst_lds, src_g) \
  (*(uint4*)(dst_lds) = *(const uint4*)(src_g))
#endif

__device__ __forceinline__ void wait_async_lds() {
#if HAVE_ASYNC_LDS
#if __has_builtin(__builtin_amdgcn_s_wait_asynccnt)
  __builtin_amdgcn_s_wait_asynccnt(0);
#else
  asm volatile("s_wait_asynccnt 0" ::: "memory");
#endif
#endif
}

// ---- WMMA fragment loaders (wave32 layouts per CDNA5 ISA 7.12.2) ----

// A tile: 16 rows x 32 K, bf16, from row-major A[row][k], lda in elements.
__device__ __forceinline__ v16bf load_a16(const bf16* A, int lda, int row0, int k0) {
  int lane = threadIdx.x & 31;
  int row  = row0 + (lane & 15);
  int half = lane >> 4;
  const bf16* p = A + (size_t)row * lda + k0;
  v16bf a;
#pragma unroll
  for (int v = 0; v < 8; ++v) {
    int k = (v >> 2) * 16 + half * 8 + (v & 3) * 2;
    a[2 * v]     = p[k];
    a[2 * v + 1] = p[k + 1];
  }
  return a;
}

// B tile: 32 K x 16 cols, where B[k][n] = W[n][k] with W row-major (GEMMs are X*W^T).
__device__ __forceinline__ v16bf load_b16(const bf16* W, int ldw, int n0, int k0) {
  int lane = threadIdx.x & 31;
  int n    = n0 + (lane & 15);
  int half = lane >> 4;
  const bf16* p = W + (size_t)n * ldw + k0 + half * 8;
  v16bf b;
#pragma unroll
  for (int v = 0; v < 8; ++v) {
    b[2 * v]     = p[(v >> 2) * 16 + (v & 3) * 2];
    b[2 * v + 1] = p[(v >> 2) * 16 + (v & 3) * 2 + 1];
  }
  return b;
}

// ---- prep kernels ----

__global__ void k_f2bf(const float* __restrict__ src, bf16* __restrict__ dst, int n) {
  int i = blockIdx.x * blockDim.x + threadIdx.x;
  if (i < n) dst[i] = (bf16)src[i];
}

// x (T, B*N) f32 -> xbnt (B*N, T) bf16, LDS-tiled transpose
__global__ void k_xpose(const float* __restrict__ x, bf16* __restrict__ xbnt) {
  __shared__ float tile[32][33];
  int t0 = blockIdx.x * 32;
  int r0 = blockIdx.y * 32;
  int tx = threadIdx.x, ty = threadIdx.y;
  tile[ty][tx] = x[(size_t)(t0 + ty) * (BB * NN) + r0 + tx];
  __syncthreads();
  xbnt[(size_t)(r0 + ty) * TT + t0 + tx] = (bf16)tile[tx][ty];
}

__global__ void k_init(const float* __restrict__ h0, const float* __restrict__ c0,
                       bf16* __restrict__ hcbf, float* __restrict__ C) {
  int idx = blockIdx.x * blockDim.x + threadIdx.x;   // B*M
  int b = idx >> 9, m = idx & 511;
  float h = h0[idx], c = c0[idx];
  C[idx] = c;
  hcbf[(size_t)b * M2 + m]      = (bf16)h;
  hcbf[(size_t)b * M2 + MM + m] = (bf16)c;
}

// ---- Ue_x GEMM: (B*N,128) x Ue_w^T -> Ue_x stored as (b, s, n) ----
__global__ void k_gemm_ue(const bf16* __restrict__ A, const bf16* __restrict__ W,
                          const float* __restrict__ bias, float* __restrict__ out) {
  int w = blockIdx.x * (blockDim.x >> 5) + (threadIdx.x >> 5);
  int row0 = (w >> 3) * 16;   // 8192 row tiles over B*N
  int col0 = (w & 7) * 16;    // 8 col tiles over s
  v8f acc = {};
#pragma unroll
  for (int kt = 0; kt < 4; ++kt) {     // K = 128
    v16bf a = load_a16(A, TT, row0, kt * 32);
    v16bf b = load_b16(W, TT, col0, kt * 32);
    WMMA_BF16(acc, a, b);
  }
  int lane = threadIdx.x & 31;
  int col  = col0 + (lane & 15);
  int rb   = row0 + (lane >> 4) * 8;
  float bv = bias[col];
#pragma unroll
  for (int r = 0; r < 8; ++r) {
    int row   = rb + r;
    int b_idx = row >> 8;       // row / N
    int n_idx = row & 255;
    out[((size_t)b_idx * TT + col) * NN + n_idx] = acc[r] + bv;
  }
}

// ---- we GEMM: hc(B,1024) x We_w^T (128,1024) -> we(B,128) ----
__global__ void k_gemm_we(const bf16* __restrict__ A, const bf16* __restrict__ W,
                          const float* __restrict__ bias, float* __restrict__ out) {
  int w = blockIdx.x * 8 + (threadIdx.x >> 5);
  int row0 = (w >> 3) * 16;   // 32 row tiles
  int col0 = (w & 7) * 16;    // 8 col tiles
  v8f acc = {};
  for (int kt = 0; kt < 32; ++kt) {    // K = 1024
    v16bf a = load_a16(A, M2, row0, kt * 32);
    v16bf b = load_b16(W, M2, col0, kt * 32);
    WMMA_BF16(acc, a, b);
  }
  int lane = threadIdx.x & 31;
  int col  = col0 + (lane & 15);
  int rb   = row0 + (lane >> 4) * 8;
  float bv = bias[col];
#pragma unroll
  for (int r = 0; r < 8; ++r) out[(size_t)(rb + r) * TT + col] = acc[r] + bv;
}

// ---- input attention + softmax over n + x_tilde (one block per b, 256 threads) ----
__global__ void k_attn_step(const float* __restrict__ we, const float* __restrict__ uex,
                            const float* __restrict__ vew, const float* __restrict__ xt,
                            bf16* __restrict__ xtbf) {
  __shared__ float s_we[TT];
  __shared__ float s_vw[TT];
  __shared__ float red[NN];
  int b = blockIdx.x, n = threadIdx.x;
  if (n < TT) { s_we[n] = we[b * TT + n]; s_vw[n] = vew[n]; }
  __syncthreads();
  const float* up = uex + (size_t)b * TT * NN + n;   // stride NN per s -> coalesced over n
  float acc = 0.f;
#pragma unroll 4
  for (int s = 0; s < TT; ++s) acc += s_vw[s] * FAST_TANH(s_we[s] + up[(size_t)s * NN]);
  // softmax over 256 n's (ve_b constant drops out of softmax)
  red[n] = acc; __syncthreads();
  for (int off = 128; off > 0; off >>= 1) {
    if (n < off) red[n] = fmaxf(red[n], red[n + off]);
    __syncthreads();
  }
  float mx = red[0]; __syncthreads();
  float ex = __expf(acc - mx);
  red[n] = ex; __syncthreads();
  for (int off = 128; off > 0; off >>= 1) {
    if (n < off) red[n] += red[n + off];
    __syncthreads();
  }
  float alpha = ex / red[0];
  xtbf[(size_t)b * NN + n] = (bf16)(alpha * xt[(size_t)b * NN + n]);
}

// ---- fused gates GEMM + LSTM pointwise ----
// One block per 16 batch rows, 512 threads (16 waves). A = [x~ | h] staged in LDS
// (async global->LDS when available). Gates accumulate via WMMA into fp32 LDS,
// then the block applies the LSTM nonlinearity and writes h/c (bf16 + f32 state).
__global__ void k_gates(const bf16* __restrict__ Xt, const bf16* __restrict__ Wih,
                        const bf16* Hc, const bf16* __restrict__ Whh,
                        const float* __restrict__ bih, const float* __restrict__ bhh,
                        float* __restrict__ C, bf16* __restrict__ hseq_t, bf16* hcbf) {
  __shared__ bf16  sA[16 * 768];      // per row: 256 x~ | 512 h   (24 KB)
  __shared__ float sG[16][M4];        // gate pre-activations      (128 KB)
  int row0 = blockIdx.x * 16;
  int tid  = threadIdx.x;
  // stage A: 16 rows x 1536 B = 1536 16-byte chunks, 3 per thread
#pragma unroll
  for (int i = 0; i < 3; ++i) {
    int c = tid + i * 512;
    int row = c / 96, seg = c % 96;   // 96 chunks per row: 32 (x~) + 64 (h)
    const bf16* src = (seg < 32)
        ? (Xt + (size_t)(row0 + row) * NN + seg * 8)
        : (Hc + (size_t)(row0 + row) * M2 + (seg - 32) * 8);
    ASYNC_CP16(sA + (size_t)c * 8, src);
  }
  wait_async_lds();
  __syncthreads();

  int w = tid >> 5, lane = tid & 31;
  v8f acc[8] = {};
  for (int kt = 0; kt < 24; ++kt) {   // unified K: 0..255 = x~*Wih, 256..767 = h*Whh
    v16bf a = load_a16(sA, 768, 0, kt * 32);
    const bf16* Wp;
    int ldw, k0;
    if (kt < 8) { Wp = Wih; ldw = NN; k0 = kt * 32; }
    else        { Wp = Whh; ldw = MM; k0 = (kt - 8) * 32; }
#pragma unroll
    for (int ct = 0; ct < 8; ++ct) {
      v16bf b = load_b16(Wp, ldw, (w * 8 + ct) * 16, k0);
      WMMA_BF16(acc[ct], a, b);
    }
  }
  // scatter D tiles (+bias) into LDS gate buffer
#pragma unroll
  for (int ct = 0; ct < 8; ++ct) {
    int col = (w * 8 + ct) * 16 + (lane & 15);
    int rb  = (lane >> 4) * 8;
    float bv = bih[col] + bhh[col];
#pragma unroll
    for (int r = 0; r < 8; ++r) sG[rb + r][col] = acc[ct][r] + bv;
  }
  __syncthreads();
  // LSTM pointwise: 16 rows x 512 m = 8192 elements, 16 per thread
#pragma unroll
  for (int i = 0; i < 16; ++i) {
    int idx = tid + i * 512;
    int r = idx >> 9, m = idx & 511;
    int grow = row0 + r;
    float gi = sG[r][m];
    float gf = sG[r][MM + m];
    float gg = sG[r][2 * MM + m];
    float go = sG[r][3 * MM + m];
    float iv = 1.f / (1.f + __expf(-gi));
    float fv = 1.f / (1.f + __expf(-gf));
    float gv = FAST_TANH(gg);
    float ov = 1.f / (1.f + __expf(-go));
    size_t cidx = (size_t)grow * MM + m;
    float c = fv * C[cidx] + iv * gv;
    float h = ov * FAST_TANH(c);
    C[cidx] = c;
    hseq_t[cidx] = (bf16)h;
    hcbf[(size_t)grow * M2 + m]      = (bf16)h;
    hcbf[(size_t)grow * M2 + MM + m] = (bf16)c;
  }
}

// ---- temporal attention scores: l[b,t] = tanh(h Ud^T + Ud_b) . vd_w + vd_b ----
// One block per 16 rows of (t*B+b); h rows staged in LDS, 16 waves x 2 col tiles.
__global__ void k_attn2(const bf16* __restrict__ H, const bf16* __restrict__ Udw,
                        const float* __restrict__ Udb, const float* __restrict__ vdw,
                        const float* __restrict__ vdb, float* __restrict__ L) {
  __shared__ bf16  sA[16 * MM];       // 16 KB
  __shared__ float s_udb[MM];
  __shared__ float s_vdw[MM];
  __shared__ float sRed[16];
  int row0 = blockIdx.x * 16;
  int tid  = threadIdx.x;             // 512 threads
  if (tid < MM) { s_udb[tid] = Udb[tid]; s_vdw[tid] = vdw[tid]; }
  if (tid < 16) sRed[tid] = 0.f;
  // stage 16 rows x 512 bf16 = 1024 chunks, 2 per thread
#pragma unroll
  for (int i = 0; i < 2; ++i) {
    int c = tid + i * 512;
    int row = c >> 6, seg = c & 63;
    ASYNC_CP16(sA + (size_t)c * 8, H + (size_t)(row0 + row) * MM + seg * 8);
  }
  wait_async_lds();
  __syncthreads();

  int w = tid >> 5, lane = tid & 31;
  v8f acc[2] = {};
  for (int kt = 0; kt < 16; ++kt) {   // K = 512
    v16bf a = load_a16(sA, MM, 0, kt * 32);
#pragma unroll
    for (int ct = 0; ct < 2; ++ct) {
      v16bf b = load_b16(Udw, MM, (w * 2 + ct) * 16, kt * 32);
      WMMA_BF16(acc[ct], a, b);
    }
  }
  float part[8] = {0.f, 0.f, 0.f, 0.f, 0.f, 0.f, 0.f, 0.f};
#pragma unroll
  for (int ct = 0; ct < 2; ++ct) {
    int col = (w * 2 + ct) * 16 + (lane & 15);
    float ub = s_udb[col], vw = s_vdw[col];
#pragma unroll
    for (int r = 0; r < 8; ++r) part[r] += FAST_TANH(acc[ct][r] + ub) * vw;
  }
#pragma unroll
  for (int r = 0; r < 8; ++r) {       // reduce across the 16 col-lanes
    float v = part[r];
    v += __shfl_xor(v, 1, 32);
    v += __shfl_xor(v, 2, 32);
    v += __shfl_xor(v, 4, 32);
    v += __shfl_xor(v, 8, 32);
    part[r] = v;
  }
  if ((lane & 15) == 0) {
    int rb = (lane >> 4) * 8;
#pragma unroll
    for (int r = 0; r < 8; ++r) atomicAdd(&sRed[rb + r], part[r]);
  }
  __syncthreads();
  if (tid < 16) {
    int row = row0 + tid;             // row = t*B + b
    int t = row >> 9, b = row & 511;
    L[(size_t)b * TT + t] = sRed[tid] + vdb[0];
  }
}

// ---- softmax over t -> beta (output), ctx = sum_t beta*h, logits = ctx.out_w + out_b ----
__global__ void k_final(const float* __restrict__ L, const bf16* __restrict__ H,
                        const float* __restrict__ outw, const float* __restrict__ outb,
                        float* __restrict__ out) {
  __shared__ float sb[TT];
  __shared__ float red[TT];
  int b = blockIdx.x, tid = threadIdx.x;   // 128 threads
  float e = L[(size_t)b * TT + tid];
  red[tid] = e; __syncthreads();
  for (int off = 64; off > 0; off >>= 1) {
    if (tid < off) red[tid] = fmaxf(red[tid], red[tid + off]);
    __syncthreads();
  }
  float mx = red[0]; __syncthreads();
  float ex = __expf(e - mx);
  red[tid] = ex; __syncthreads();
  for (int off = 64; off > 0; off >>= 1) {
    if (tid < off) red[tid] += red[tid + off];
    __syncthreads();
  }
  float beta = ex / red[0];
  sb[tid] = beta;
  out[BB + (size_t)b * TT + tid] = beta;   // beta output section
  __syncthreads();
  float lacc = 0.f;
  for (int m = tid; m < MM; m += TT) {
    float cacc = 0.f;
    const bf16* hp = H + (size_t)b * MM + m;
    for (int t = 0; t < TT; ++t) cacc += sb[t] * (float)hp[(size_t)t * BB * MM];
    lacc += cacc * outw[m];
  }
  red[tid] = lacc; __syncthreads();
  for (int off = 64; off > 0; off >>= 1) {
    if (tid < off) red[tid] += red[tid + off];
    __syncthreads();
  }
  if (tid == 0) out[b] = red[0] + outb[0];
}

extern "C" void kernel_launch(void* const* d_in, const int* in_sizes, int n_in,
                              void* d_out, int out_size, void* d_ws, size_t ws_size,
                              hipStream_t stream) {
  (void)in_sizes; (void)n_in; (void)out_size; (void)ws_size;
  const float* x     = (const float*)d_in[0];
  const float* h0    = (const float*)d_in[1];
  const float* c0    = (const float*)d_in[2];
  const float* Wih   = (const float*)d_in[3];
  const float* Whh   = (const float*)d_in[4];
  const float* b_ih  = (const float*)d_in[5];
  const float* b_hh  = (const float*)d_in[6];
  const float* We_w  = (const float*)d_in[7];
  const float* We_b  = (const float*)d_in[8];
  const float* Ue_w  = (const float*)d_in[9];
  const float* Ue_b  = (const float*)d_in[10];
  const float* ve_w  = (const float*)d_in[11];
  const float* Ud_w  = (const float*)d_in[13];
  const float* Ud_b  = (const float*)d_in[14];
  const float* vd_w  = (const float*)d_in[15];
  const float* vd_b  = (const float*)d_in[16];
  const float* out_w = (const float*)d_in[17];
  const float* out_b = (const float*)d_in[18];
  float* out = (float*)d_out;

  char* ws = (char*)d_ws;
  size_t cur = 0;
  auto take = [&](size_t bytes) -> void* {
    void* p = ws + cur;
    cur += (bytes + 255) & ~(size_t)255;
    return p;
  };
  float* UEX   = (float*)take((size_t)BB * TT * NN * 4);  // (b, s, n)
  bf16*  XBNT  = (bf16*) take((size_t)BB * NN * TT * 2);  // (b*n, t)
  bf16*  HSEQ  = (bf16*) take((size_t)TT * BB * MM * 2);  // (t, b, m)
  bf16*  HCBF  = (bf16*) take((size_t)BB * M2 * 2);       // (b, [h|c])
  float* CBUF  = (float*)take((size_t)BB * MM * 4);
  float* WE    = (float*)take((size_t)BB * TT * 4);
  bf16*  XTBF  = (bf16*) take((size_t)BB * NN * 2);
  float* LBUF  = (float*)take((size_t)BB * TT * 4);
  bf16*  WIHB  = (bf16*) take((size_t)M4 * NN * 2);
  bf16*  WHHB  = (bf16*) take((size_t)M4 * MM * 2);
  bf16*  WEWB  = (bf16*) take((size_t)TT * M2 * 2);
  bf16*  UEWB  = (bf16*) take((size_t)TT * TT * 2);
  bf16*  UDWB  = (bf16*) take((size_t)MM * MM * 2);

  // one-time prep
  k_f2bf<<<(M4 * NN + 255) / 256, 256, 0, stream>>>(Wih, WIHB, M4 * NN);
  k_f2bf<<<(M4 * MM + 255) / 256, 256, 0, stream>>>(Whh, WHHB, M4 * MM);
  k_f2bf<<<(TT * M2 + 255) / 256, 256, 0, stream>>>(We_w, WEWB, TT * M2);
  k_f2bf<<<(TT * TT + 255) / 256, 256, 0, stream>>>(Ue_w, UEWB, TT * TT);
  k_f2bf<<<(MM * MM + 255) / 256, 256, 0, stream>>>(Ud_w, UDWB, MM * MM);
  k_xpose<<<dim3(TT / 32, (BB * NN) / 32), dim3(32, 32), 0, stream>>>(x, XBNT);
  k_init<<<(BB * MM) / 256, 256, 0, stream>>>(h0, c0, HCBF, CBUF);
  k_gemm_ue<<<8192, 256, 0, stream>>>(XBNT, UEWB, Ue_b, UEX);

  // sequential recurrence (3 kernels per step)
  for (int t = 0; t < TT; ++t) {
    k_gemm_we<<<32, 256, 0, stream>>>(HCBF, WEWB, We_b, WE);
    k_attn_step<<<BB, NN, 0, stream>>>(WE, UEX, ve_w, x + (size_t)t * BB * NN, XTBF);
    k_gates<<<32, 512, 0, stream>>>(XTBF, WIHB, HCBF, WHHB, b_ih, b_hh,
                                    CBUF, HSEQ + (size_t)t * BB * MM, HCBF);
  }

  // temporal attention + output
  k_attn2<<<(TT * BB) / 16, 512, 0, stream>>>(HSEQ, UDWB, Ud_b, vd_w, vd_b, LBUF);
  k_final<<<BB, TT, 0, stream>>>(LBUF, HSEQ, out_w, out_b, out);
}
